// MultiScaleHybridREWAEncoder_67534065762856
// MI455X (gfx1250) — matrix operand order repaired and checked by reading, not compiled, not run on hardware
//
#include <hip/hip_runtime.h>
#include <hip/hip_bf16.h>
#include <math.h>

// ---------------------------------------------------------------------------
// MultiScaleHybridREWAEncoder for MI455X (gfx1250, wave32, WMMA)
//
// Fused single-pass design: x (f32) is read once, converted to bf16, and the
// entire per-token pipeline (4 projections -> norms -> q/k/v -> 4-way softmax
// attention -> out-proj -> L2) runs inside one workgroup using ~288KB of the
// 320KB/WGP LDS. All GEMMs use v_wmma_f32_16x16x32_bf16 (f32 accumulate).
// Weights are pre-packed into WMMA B-operand tile layout in d_ws (~2MB).
//
// Round-2 changes: B tiles preloaded into register arrays per K-step so the
// scheduler can use partial s_wait_loadcnt instead of full drains; stage-2
// operand preloading; float4 LDS sweeps in the normalization passes.
// ---------------------------------------------------------------------------

typedef __attribute__((ext_vector_type(16))) __bf16 v16bf;
typedef __attribute__((ext_vector_type(8)))  float  v8f;

#define D_MODEL 768
#define M_DIM   256
#define NBIG    1024        // 4 branches * 256
#define TM      32          // tokens per workgroup
#define KT1     24          // 768/32 k-tiles, stage 1
#define KT2     8           // 256/32 k-tiles, stage 2

// ---- dynamic LDS layout (bytes), phased reuse, total 294912 <= 320KB ----
#define OFF_OUT1    0        // f32 [32][1024]            (phase A..norm)
#define OFF_XA      131072   // bf16 A-layout 2*24*512    (phase A)
#define OFF_FEATSA  147456   // bf16 A-layout 8*8*512     (overlays dead XA)
#define OFF_QA      212992   // bf16 A-layout 2*8*512
#define OFF_QH      0        // f32 [32][256]             (overlays dead OUT1)
#define OFF_KHB     32768    // bf16 [128][256]
#define OFF_VHB     229376   // bf16 [128][256]
#define OFF_FUSEDA  147456   // bf16 A-layout 2*8*512     (overlays dead FEATSA)
#define OFF_OUTACC  0        // f32 [32][256]             (overlays dead QH)
#define OFF_SCALE   163840   // f32 [32]
#define LDS_BYTES   294912

__device__ __forceinline__ unsigned short f2bf(float f) {
  unsigned u = __float_as_uint(f);
  return (unsigned short)((u + 0x7FFFu + ((u >> 16) & 1u)) >> 16); // RNE
}
__device__ __forceinline__ float bf2f(unsigned short b) {
  return __uint_as_float(((unsigned)b) << 16);
}

// A-operand (16-bit 16x32) slot->k mapping per ISA 7.12.2:
//   lane<16 : slots 0-7 -> K 0-7,  slots 8-15 -> K 16-23
//   lane>=16: slots 0-7 -> K 8-15, slots 8-15 -> K 24-31
__device__ __forceinline__ int a_k_of(int ln, int sl) {
  return ((sl >> 3) << 4) + ((ln >> 4) << 3) + (sl & 7);
}

// ---------------------------------------------------------------------------
// Pack a torch-layout weight W[N][K] (row-major f32) into WMMA B-tile layout:
// dst[kt][nt][lane][slot] bf16, where col = lane&15, k = kt*32 + (slot>>3)*16
// + (lane>>4)*8 + (slot&7) (same slot->k mapping as the A operand, a_k_of).
// ---------------------------------------------------------------------------
__global__ __launch_bounds__(256)
void pack_b_kernel(const float* __restrict__ W, unsigned short* __restrict__ dst,
                   int N, int K) {
  int idx = blockIdx.x * 256 + threadIdx.x;
  int total = (K >> 5) * (N >> 4) * 512;
  if (idx >= total) return;
  int sl = idx & 15;
  int ln = (idx >> 4) & 31;
  int nt = (idx >> 9) % (N >> 4);
  int kt = (idx >> 9) / (N >> 4);
  int n = nt * 16 + (ln & 15);
  int k = kt * 32 + ((sl >> 3) << 4) + ((ln >> 4) << 3) + (sl & 7);
  dst[idx] = f2bf(W[n * K + k]);
}

// ---------------------------------------------------------------------------
// Main fused kernel: one workgroup = 32 tokens, 256 threads = 8 wave32.
// ---------------------------------------------------------------------------
__global__ __launch_bounds__(256)
void rewa_main_kernel(const float* __restrict__ x,
                      const float* __restrict__ bl, const float* __restrict__ ln_g,
                      const float* __restrict__ ln_b,
                      const float* __restrict__ bq, const float* __restrict__ bk,
                      const float* __restrict__ bv, const float* __restrict__ bo,
                      const unsigned short* __restrict__ wbig, // [4][24][16][512]
                      const unsigned short* __restrict__ wq,   // [8][16][512]
                      const unsigned short* __restrict__ wk,
                      const unsigned short* __restrict__ wv,
                      const unsigned short* __restrict__ wo,
                      float* __restrict__ out) {
  extern __shared__ char smem[];
  float*          out1   = (float*)(smem + OFF_OUT1);
  unsigned short* xA     = (unsigned short*)(smem + OFF_XA);
  unsigned short* featsA = (unsigned short*)(smem + OFF_FEATSA);
  unsigned short* qA     = (unsigned short*)(smem + OFF_QA);
  float*          qh     = (float*)(smem + OFF_QH);
  unsigned short* khb    = (unsigned short*)(smem + OFF_KHB);
  unsigned short* vhb    = (unsigned short*)(smem + OFF_VHB);
  unsigned short* fusedA = (unsigned short*)(smem + OFF_FUSEDA);
  float*          outacc = (float*)(smem + OFF_OUTACC);
  float*          scaleA = (float*)(smem + OFF_SCALE);

  const int tid  = threadIdx.x;
  const int wave = tid >> 5;
  const int lane = tid & 31;
  const long tok0 = (long)blockIdx.x * TM;

  // ---- Phase A: x tile f32 -> bf16, written directly in A-operand layout ----
  for (int i = tid; i < 2 * KT1 * 512; i += 256) {
    int blk = i >> 9;                 // rt*24 + kt
    int rt = blk / KT1, kt = blk % KT1;
    int ln = (i >> 4) & 31, sl = i & 15;
    int row = rt * 16 + (ln & 15);
    int k = kt * 32 + a_k_of(ln, sl);
    xA[i] = f2bf(x[(tok0 + row) * D_MODEL + k]);
  }
  __syncthreads();

  // ---- Stage 1: [32x768] x [768x1024] -> out1 (f32, LDS) --------------------
  {
    // This wave's 8 column tiles are contiguous in the packed layout:
    // branch = wave>>1 (uniform per wave), first in-branch tile = (wave&1)*8.
    const unsigned short* wwave =
        wbig + (size_t)(wave >> 1) * (KT1 * 16 * 512) + (size_t)((wave & 1) * 8) * 512;
    v8f acc[16] = {};
    for (int kt = 0; kt < KT1; ++kt) {
      v16bf a0 = *(const v16bf*)(xA + (0 * KT1 + kt) * 512 + lane * 16);
      v16bf a1 = *(const v16bf*)(xA + (1 * KT1 + kt) * 512 + lane * 16);
      v16bf bm[8];
#pragma unroll
      for (int j = 0; j < 8; ++j)
        bm[j] = *(const v16bf*)(wwave + (size_t)kt * (16 * 512) + j * 512 + lane * 16);
#pragma unroll
      for (int j = 0; j < 8; ++j) {
        acc[j]     = __builtin_amdgcn_wmma_f32_16x16x32_bf16(false, a0, false, bm[j], (short)0, acc[j],     false, false);
        acc[8 + j] = __builtin_amdgcn_wmma_f32_16x16x32_bf16(false, a1, false, bm[j], (short)0, acc[8 + j], false, false);
      }
    }
    // C/D layout: lane<16 -> M=v, lane>=16 -> M=8+v; N = lane&15 within tile.
#pragma unroll
    for (int j = 0; j < 8; ++j) {
      int col = (wave * 8 + j) * 16 + (lane & 15);
      int r0 = ((lane >> 4) << 3);
#pragma unroll
      for (int v = 0; v < 8; ++v) {
        out1[(r0 + v) * NBIG + col]      = acc[j][v];
        out1[(16 + r0 + v) * NBIG + col] = acc[8 + j][v];
      }
    }
  }
  __syncthreads();

  // ---- per-(token,branch) normalization (branch 3: +bl, LayerNorm, L2) -----
  if (tid < 128) {
    int t = tid >> 2, b = tid & 3;
    float* row = out1 + t * NBIG + b * M_DIM;
    float4* row4 = (float4*)row;
    if (b < 3) {
      float ss = 0.f;
      for (int j = 0; j < M_DIM / 4; ++j) {
        float4 v = row4[j];
        ss += v.x * v.x + v.y * v.y + v.z * v.z + v.w * v.w;
      }
      float sc = 1.f / fmaxf(sqrtf(ss), 1e-12f);
      for (int j = 0; j < M_DIM / 4; ++j) {
        float4 v = row4[j];
        v.x *= sc; v.y *= sc; v.z *= sc; v.w *= sc;
        row4[j] = v;
      }
    } else {
      const float4* bl4 = (const float4*)bl;
      float s = 0.f, ss = 0.f;
      for (int j = 0; j < M_DIM / 4; ++j) {
        float4 v = row4[j]; float4 bb = bl4[j];
        v.x += bb.x; v.y += bb.y; v.z += bb.z; v.w += bb.w;
        s  += v.x + v.y + v.z + v.w;
        ss += v.x * v.x + v.y * v.y + v.z * v.z + v.w * v.w;
        row4[j] = v;
      }
      float mu = s * (1.f / M_DIM);
      float var = ss * (1.f / M_DIM) - mu * mu;
      float inv = rsqrtf(var + 1e-5f);
      const float4* g4 = (const float4*)ln_g;
      const float4* b4 = (const float4*)ln_b;
      float l2 = 0.f;
      for (int j = 0; j < M_DIM / 4; ++j) {
        float4 v = row4[j]; float4 gg = g4[j]; float4 be = b4[j];
        v.x = (v.x - mu) * inv * gg.x + be.x;
        v.y = (v.y - mu) * inv * gg.y + be.y;
        v.z = (v.z - mu) * inv * gg.z + be.z;
        v.w = (v.w - mu) * inv * gg.w + be.w;
        l2 += v.x * v.x + v.y * v.y + v.z * v.z + v.w * v.w;
        row4[j] = v;
      }
      float sc = 1.f / fmaxf(sqrtf(l2), 1e-12f);
      for (int j = 0; j < M_DIM / 4; ++j) {
        float4 v = row4[j];
        v.x *= sc; v.y *= sc; v.z *= sc; v.w *= sc;
        row4[j] = v;
      }
    }
  }
  __syncthreads();

  // ---- pack feats (128 rows) and q = branch-mean (32 rows) as bf16 A tiles --
  for (int i = tid; i < 8 * KT2 * 512; i += 256) {
    int blk = i >> 9; int rt = blk >> 3, kt = blk & 7;
    int ln = (i >> 4) & 31, sl = i & 15;
    int fr = rt * 16 + (ln & 15);                 // feat row = t*4 + b
    int k = kt * 32 + a_k_of(ln, sl);
    featsA[i] = f2bf(out1[(fr >> 2) * NBIG + (fr & 3) * M_DIM + k]);
  }
  for (int i = tid; i < 2 * KT2 * 512; i += 256) {
    int blk = i >> 9; int rt = blk >> 3, kt = blk & 7;
    int ln = (i >> 4) & 31, sl = i & 15;
    int t = rt * 16 + (ln & 15);
    int k = kt * 32 + a_k_of(ln, sl);
    const float* rr = out1 + t * NBIG;
    qA[i] = f2bf(0.25f * (rr[k] + rr[256 + k] + rr[512 + k] + rr[768 + k]));
  }
  __syncthreads();

  // ---- Stage 2: kh (128 tiles) / vh (128) / qh (32) projections -------------
  for (int tile = wave; tile < 288; tile += 8) {
    int mode, rt, nt;
    const unsigned short* W;
    if (tile < 128)      { mode = 0; W = wk; rt = tile >> 4;         nt = tile & 15; }
    else if (tile < 256) { mode = 1; W = wv; rt = (tile - 128) >> 4; nt = tile & 15; }
    else                 { mode = 2; W = wq; rt = (tile - 256) >> 4; nt = tile & 15; }
    const unsigned short* Abase = (mode == 2) ? qA : featsA;
    // preload all operands, then run the 8-step WMMA chain
    v16bf aArr[KT2], bArr[KT2];
#pragma unroll
    for (int kt = 0; kt < KT2; ++kt) {
      aArr[kt] = *(const v16bf*)(Abase + (rt * KT2 + kt) * 512 + lane * 16);
      bArr[kt] = *(const v16bf*)(W + (kt * 16 + nt) * 512 + lane * 16);
    }
    v8f c = {};
#pragma unroll
    for (int kt = 0; kt < KT2; ++kt)
      c = __builtin_amdgcn_wmma_f32_16x16x32_bf16(false, aArr[kt], false, bArr[kt], (short)0, c, false, false);
    int col = nt * 16 + (lane & 15);
    float bias = (mode == 0) ? bk[col] : (mode == 1) ? bv[col] : bq[col];
    int r0 = ((lane >> 4) << 3);
    if (mode == 2) {
#pragma unroll
      for (int v = 0; v < 8; ++v) qh[(rt * 16 + r0 + v) * M_DIM + col] = c[v] + bias;
    } else {
      unsigned short* dst = (mode == 0) ? khb : vhb;
#pragma unroll
      for (int v = 0; v < 8; ++v) dst[(rt * 16 + r0 + v) * M_DIM + col] = f2bf(c[v] + bias);
    }
  }
  __syncthreads();

  // ---- 4-way softmax attention per (token, head); write fused as A tiles ----
  if (tid < 128) {
    int t = tid >> 2, h = tid & 3;
    const float* qrow = qh + t * M_DIM + h * 64;
    float sc[4];
#pragma unroll
    for (int kb = 0; kb < 4; ++kb) {
      const unsigned short* krow = khb + (t * 4 + kb) * M_DIM + h * 64;
      float s = 0.f;
      for (int d = 0; d < 64; ++d) s += qrow[d] * bf2f(krow[d]);
      sc[kb] = s * 0.125f;                         // 1/sqrt(64)
    }
    float mx = fmaxf(fmaxf(sc[0], sc[1]), fmaxf(sc[2], sc[3]));
    float e0 = __expf(sc[0] - mx), e1 = __expf(sc[1] - mx);
    float e2 = __expf(sc[2] - mx), e3 = __expf(sc[3] - mx);
    float inv = 1.f / (e0 + e1 + e2 + e3);
    e0 *= inv; e1 *= inv; e2 *= inv; e3 *= inv;
    for (int d = 0; d < 64; ++d) {
      int col = h * 64 + d;
      float f = e0 * bf2f(vhb[(t * 4 + 0) * M_DIM + col])
              + e1 * bf2f(vhb[(t * 4 + 1) * M_DIM + col])
              + e2 * bf2f(vhb[(t * 4 + 2) * M_DIM + col])
              + e3 * bf2f(vhb[(t * 4 + 3) * M_DIM + col]);
      // inverse of a_k_of: place (row=t, k=col) into A-operand layout
      int rt = t >> 4, kt = col >> 5, kk = col & 31;
      int ln = (t & 15) + (((kk >> 3) & 1) << 4);
      int sl = ((kk >> 4) << 3) | (kk & 7);
      fusedA[(rt * KT2 + kt) * 512 + ln * 16 + sl] = f2bf(f);
    }
  }
  __syncthreads();

  // ---- out projection: [32x256] x Wo^T --------------------------------------
  for (int tile = wave; tile < 32; tile += 8) {
    int rt = tile >> 4, nt = tile & 15;
    v16bf aArr[KT2], bArr[KT2];
#pragma unroll
    for (int kt = 0; kt < KT2; ++kt) {
      aArr[kt] = *(const v16bf*)(fusedA + (rt * KT2 + kt) * 512 + lane * 16);
      bArr[kt] = *(const v16bf*)(wo + (kt * 16 + nt) * 512 + lane * 16);
    }
    v8f c = {};
#pragma unroll
    for (int kt = 0; kt < KT2; ++kt)
      c = __builtin_amdgcn_wmma_f32_16x16x32_bf16(false, aArr[kt], false, bArr[kt], (short)0, c, false, false);
    int col = nt * 16 + (lane & 15);
    float bias = bo[col];
    int r0 = ((lane >> 4) << 3);
#pragma unroll
    for (int v = 0; v < 8; ++v) outacc[(rt * 16 + r0 + v) * M_DIM + col] = c[v] + bias;
  }
  __syncthreads();

  // ---- final L2 norm and store ----------------------------------------------
  if (tid < TM) {
    const float4* row4 = (const float4*)(outacc + tid * M_DIM);
    float ss = 0.f;
    for (int j = 0; j < M_DIM / 4; ++j) {
      float4 v = row4[j];
      ss += v.x * v.x + v.y * v.y + v.z * v.z + v.w * v.w;
    }
    scaleA[tid] = 1.f / fmaxf(sqrtf(ss), 1e-12f);
  }
  __syncthreads();
  for (int i = tid; i < TM * M_DIM; i += 256) {
    int t = i >> 8, c = i & 255;
    out[(tok0 + t) * M_DIM + c] = outacc[t * M_DIM + c] * scaleA[t];
  }
}

// ---------------------------------------------------------------------------
// d_in order: x, Wr1, Wr2, Wr3, Wl, bl, ln_g, ln_b, Wq, bq, Wk, bk, Wv, bv, Wo, bo
// d_ws usage: 2MB of bf16 packed weights.
// ---------------------------------------------------------------------------
extern "C" void kernel_launch(void* const* d_in, const int* in_sizes, int n_in,
                              void* d_out, int out_size, void* d_ws, size_t ws_size,
                              hipStream_t stream) {
  (void)in_sizes; (void)n_in; (void)out_size; (void)ws_size;
  const float* x    = (const float*)d_in[0];
  const float* bl   = (const float*)d_in[5];
  const float* ln_g = (const float*)d_in[6];
  const float* ln_b = (const float*)d_in[7];
  const float* bq   = (const float*)d_in[9];
  const float* bk   = (const float*)d_in[11];
  const float* bv   = (const float*)d_in[13];
  const float* bo   = (const float*)d_in[15];

  unsigned short* ws   = (unsigned short*)d_ws;
  unsigned short* wbig = ws;                         // 4 * 196608 bf16
  unsigned short* wqp  = ws + 4 * 196608;            // 65536 bf16 each
  unsigned short* wkp  = wqp + 65536;
  unsigned short* wvp  = wkp + 65536;
  unsigned short* wop  = wvp + 65536;

  // pack stage-1 branch weights [256x768] (Wr1, Wr2, Wr3, Wl)
  for (int b = 0; b < 4; ++b) {
    const float* Wsrc = (const float*)d_in[1 + b];
    pack_b_kernel<<<768, 256, 0, stream>>>(Wsrc, wbig + (size_t)b * 196608, M_DIM, D_MODEL);
  }
  // pack stage-2 weights [256x256]
  pack_b_kernel<<<256, 256, 0, stream>>>((const float*)d_in[8],  wqp, M_DIM, M_DIM);
  pack_b_kernel<<<256, 256, 0, stream>>>((const float*)d_in[10], wkp, M_DIM, M_DIM);
  pack_b_kernel<<<256, 256, 0, stream>>>((const float*)d_in[12], wvp, M_DIM, M_DIM);
  pack_b_kernel<<<256, 256, 0, stream>>>((const float*)d_in[14], wop, M_DIM, M_DIM);

  // 65536 tokens / 32 per workgroup = 2048 workgroups, 288KB dynamic LDS each
  rewa_main_kernel<<<2048, 256, LDS_BYTES, stream>>>(
      x, bl, ln_g, ln_b, bq, bk, bv, bo,
      wbig, wqp, wkp, wvp, wop, (float*)d_out);
}
